// EdgeUpdateLayer_15040975470645
// MI455X (gfx1250) — matrix operand order repaired and checked by reading, not compiled, not run on hardware
//
#include <hip/hip_runtime.h>

#define NODE_DIM 128
#define EDGE_DIM 128
#define IN_DIM   384   // 2*NODE_DIM + EDGE_DIM
#define HIDDEN   256

typedef __attribute__((ext_vector_type(16))) __bf16 v16bf;
typedef __attribute__((ext_vector_type(8)))  __bf16 v8bf;
typedef __attribute__((ext_vector_type(8)))  float  v8f;
typedef __attribute__((ext_vector_type(4)))  float  v4f;

constexpr int KB1 = IN_DIM / 32;    // 12 K-blocks, GEMM1
constexpr int NB1 = HIDDEN / 16;    // 16 N-blocks, GEMM1
constexpr int KB2 = HIDDEN / 32;    // 8  K-blocks, GEMM2
constexpr int NB2 = EDGE_DIM / 16;  // 8  N-blocks, GEMM2
constexpr int W1F_ELEMS = KB1 * NB1 * 512;       // 98304 bf16 (192KB)
constexpr int W2F_ELEMS = KB2 * NB2 * 512;       // 32768 bf16 (64KB)
constexpr int WF_TOTAL  = W1F_ELEMS + W2F_ELEMS; // 131072 bf16 (256KB)
constexpr int HID_STRIDE = 40;  // padded row stride (bf16) -> 80B, 16B-aligned
constexpr int WAVES = 12;       // 384 threads/block -> 3 waves/SIMD, 1 WG/WGP

// ---------------------------------------------------------------------------
// Repack W1/W2 (f32 row-major KxN) into bf16 WMMA B-fragments, lane-interleaved:
// per (kb,nb) block of 1024B: [ lo-halves: lane*16B ][ hi-halves: lane*16B ]
// B-layout (16-bit 32x16): lane -> {half=l>>4, N=nb*16+(l&15)};
// element j -> K = kb*32 + 16*(j>>3) + 8*half + (j&7).
// ---------------------------------------------------------------------------
__global__ __launch_bounds__(256)
void pack_weights_kernel(const float* __restrict__ W1,
                         const float* __restrict__ W2,
                         __bf16* __restrict__ wf) {
  int tid = blockIdx.x * blockDim.x + threadIdx.x;
  if (tid < W1F_ELEMS) {
    int blk  = tid >> 9;                  // kb*NB1 + nb
    int e    = tid & 511;
    int part = e >> 8;                    // 0 = j<8, 1 = j>=8
    int lane = (e >> 3) & 31;
    int j    = part * 8 + (e & 7);
    int kb   = blk >> 4, nb = blk & 15;
    int half = lane >> 4, nl = lane & 15;
    int K = kb * 32 + 16 * part + 8 * half + (e & 7);
    int N = nb * 16 + nl;
    wf[tid] = (__bf16)W1[K * HIDDEN + N];
  } else if (tid < WF_TOTAL) {
    int t2   = tid - W1F_ELEMS;
    int blk  = t2 >> 9;                   // kb*NB2 + nb
    int e    = t2 & 511;
    int part = e >> 8;
    int lane = (e >> 3) & 31;
    int j    = part * 8 + (e & 7);
    int kb   = blk >> 3, nb = blk & 7;
    int half = lane >> 4, nl = lane & 15;
    int K = kb * 32 + 16 * part + 8 * half + (e & 7);
    int N = nb * 16 + nl;
    wf[tid] = (__bf16)W2[K * EDGE_DIM + N];
  }
}

// ---------------------------------------------------------------------------
// Fused kernel. Per block: stage all bf16 weight fragments into LDS (256KB),
// then each of 12 waves grid-strides over 16-edge tiles:
//   - prefetch next tile's index/edge rows (global_prefetch_b8)
//   - gather A1 fragments (bf16) straight from global
//   - pipelined: per 32-wide GEMM2 K-slab, run 2 GEMM1 N-blocks (12 WMMAs
//     each from LDS weights), ReLU+bias, bounce the 16x32 slab through the
//     per-wave LDS pad to transpose C-layout -> A-layout, then 8 GEMM2 WMMAs
//     into persistent accumulators.
//   - epilogue: + b2 + residual, f32 stores.
// ---------------------------------------------------------------------------
__global__ __launch_bounds__(WAVES * 32)
void edge_mlp_kernel(const float* __restrict__ node,
                     const float* __restrict__ edge,
                     const long long* __restrict__ eidx,
                     const float* __restrict__ b1,
                     const float* __restrict__ b2,
                     const v16bf* __restrict__ wfrag,
                     float* __restrict__ out,
                     int n_edges) {
  __shared__ __align__(32) __bf16 wsh[WF_TOTAL];                 // 256KB weights
  __shared__ __align__(32) __bf16 hid[WAVES][16][HID_STRIDE];    // 15KB bounce

  // ---- cooperative weight staging: global (L2) -> LDS, once per block
  {
    const v16bf* src = wfrag;
    v16bf* dst = (v16bf*)wsh;
    for (int i = threadIdx.x; i < WF_TOTAL / 16; i += blockDim.x) dst[i] = src[i];
  }
  __syncthreads();
  const __bf16* w1s = wsh;
  const __bf16* w2s = wsh + W1F_ELEMS;

  const int lane = threadIdx.x & 31;
  const int wave = threadIdx.x >> 5;
  const int half = lane >> 4;
  const int lr   = lane & 15;

  const int waveId      = (blockIdx.x * blockDim.x + threadIdx.x) >> 5;
  const int wavesInGrid = (gridDim.x * blockDim.x) >> 5;
  const int ntiles      = n_edges >> 4;

  for (int t = waveId; t < ntiles; t += wavesInGrid) {
    const int ebase = t << 4;
    const int erow  = ebase + lr;

    // prefetch the next tile's pointer-chase inputs while this tile computes
    const int tn = t + wavesInGrid;
    if (tn < ntiles) {
      const int erown = (tn << 4) + lr;
      __builtin_prefetch(&eidx[erown], 0, 0);
      __builtin_prefetch(&edge[(long long)erown * EDGE_DIM], 0, 0);
    }

    const long long s = eidx[erow];
    const long long d = eidx[n_edges + erow];
    const float* seg0 = node + s * NODE_DIM;
    const float* seg1 = node + d * NODE_DIM;
    const float* seg2 = edge + (long long)erow * EDGE_DIM;

    // ---- gather A1 fragments (A-layout: row=lr; j<8 -> K=kb*32+8*half+j,
    //      j>=8 -> K=kb*32+16+8*half+(j-8))
    v16bf a1[KB1];
#pragma unroll
    for (int kb = 0; kb < KB1; ++kb) {
      const float* sp = (kb < 4) ? seg0 : (kb < 8) ? seg1 : seg2;
      const int ko = (kb & 3) * 32 + half * 8;
      v4f r0 = *(const v4f*)(sp + ko);
      v4f r1 = *(const v4f*)(sp + ko + 4);
      v4f r2 = *(const v4f*)(sp + ko + 16);
      v4f r3 = *(const v4f*)(sp + ko + 20);
      v16bf a;
#pragma unroll
      for (int j = 0; j < 4; ++j) {
        a[j]      = (__bf16)r0[j];
        a[j + 4]  = (__bf16)r1[j];
        a[j + 8]  = (__bf16)r2[j];
        a[j + 12] = (__bf16)r3[j];
      }
      a1[kb] = a;
    }

    v8f acc2[NB2] = {};   // persistent GEMM2 accumulators (64 VGPRs)

    for (int kb2 = 0; kb2 < KB2; ++kb2) {
      // -- two GEMM1 N-blocks == one GEMM2 K-slab (hidden cols kb2*32..+31)
#pragma unroll
      for (int h = 0; h < 2; ++h) {
        const int nb = kb2 * 2 + h;
        v8f acc = {};
#pragma unroll
        for (int kb = 0; kb < KB1; ++kb) {
          const __bf16* bp = w1s + (kb * NB1 + nb) * 512 + lane * 8;
          union { v16bf v; struct { v8bf lo, hi; } s; } b;
          b.s.lo = *(const v8bf*)bp;
          b.s.hi = *(const v8bf*)(bp + 256);
          acc = __builtin_amdgcn_wmma_f32_16x16x32_bf16(
              false, a1[kb], false, b.v, (short)0, acc, false, false);
        }
        const float bias = b1[nb * 16 + lr];
#pragma unroll
        for (int v = 0; v < 8; ++v) {
          float x = acc[v] + bias;
          x = x > 0.0f ? x : 0.0f;
          // C-layout: col N=lr within block; VGPR v -> row M=v+8*half.
          // Hidden col within slab = h*16 + lr.
          hid[wave][v + half * 8][h * 16 + lr] = (__bf16)x;
        }
      }

      // -- transpose bounce: reload slab as A-layout fragment
      // (LDS ops are in-order within a wave; slab is wave-private)
      union { v16bf v; struct { v8bf lo, hi; } s; } a2;
      a2.s.lo = *(const v8bf*)&hid[wave][lr][half * 8];
      a2.s.hi = *(const v8bf*)&hid[wave][lr][16 + half * 8];

      // -- GEMM2: accumulate this K-slab into all 8 N-blocks
#pragma unroll
      for (int nb2 = 0; nb2 < NB2; ++nb2) {
        const __bf16* bp = w2s + (kb2 * NB2 + nb2) * 512 + lane * 8;
        union { v16bf v; struct { v8bf lo, hi; } s; } b;
        b.s.lo = *(const v8bf*)bp;
        b.s.hi = *(const v8bf*)(bp + 256);
        acc2[nb2] = __builtin_amdgcn_wmma_f32_16x16x32_bf16(
            false, a2.v, false, b.v, (short)0, acc2[nb2], false, false);
      }
    }

    // ---- epilogue: out = edge + update + b2
#pragma unroll
    for (int nb2 = 0; nb2 < NB2; ++nb2) {
      const int col = nb2 * 16 + lr;
      const float bias = b2[col];
#pragma unroll
      for (int v = 0; v < 8; ++v) {
        const int m = v + half * 8;
        const long long o = (long long)(ebase + m) * EDGE_DIM + col;
        out[o] = edge[o] + acc2[nb2][v] + bias;
      }
    }
  }
}

// ---------------------------------------------------------------------------
extern "C" void kernel_launch(void* const* d_in, const int* in_sizes, int n_in,
                              void* d_out, int out_size, void* d_ws, size_t ws_size,
                              hipStream_t stream) {
  const float*     node = (const float*)d_in[0];
  const float*     edge = (const float*)d_in[1];
  const long long* eidx = (const long long*)d_in[2];
  const float*     W1   = (const float*)d_in[3];
  const float*     b1   = (const float*)d_in[4];
  const float*     W2   = (const float*)d_in[5];
  const float*     b2   = (const float*)d_in[6];
  float*           out  = (float*)d_out;

  const int n_edges = in_sizes[1] / EDGE_DIM;
  __bf16* wf = (__bf16*)d_ws;

  // 1) repack weights into LDS-friendly bf16 fragment order
  pack_weights_kernel<<<WF_TOTAL / 256, 256, 0, stream>>>(W1, W2, wf);

  // 2) fused gather + MLP + residual (weights staged to LDS per block)
  const int blocks = 1024;   // 12 waves/block, grid-stride over 31250 tiles
  edge_mlp_kernel<<<blocks, WAVES * 32, 0, stream>>>(node, edge, eidx, b1, b2,
                                                     (const v16bf*)wf, out,
                                                     n_edges);
}
